// MultiHeadSelfAttention_73469710566144
// MI455X (gfx1250) — matrix operand order repaired and checked by reading, compiled, not run on hardware
//
#include <hip/hip_runtime.h>

// ---------------- types ----------------
typedef __attribute__((ext_vector_type(16))) __bf16 bf16x16;
typedef __attribute__((ext_vector_type(8)))  __bf16 bf16x8;
typedef __attribute__((ext_vector_type(8)))  float  f32x8;

// ---------------- problem constants ----------------
constexpr int D_MODEL = 1024;
constexpr int N_HEADS = 16;
constexpr int D_K     = 64;
constexpr int BATCH   = 4;
constexpr int SEQ     = 2048;
constexpr int M_TOK   = BATCH * SEQ;       // 8192 tokens

static __device__ __forceinline__ bf16x16 cat8(bf16x8 lo, bf16x8 hi) {
    bf16x16 r;
#pragma unroll
    for (int i = 0; i < 8; ++i) { r[i] = lo[i]; r[8 + i] = hi[i]; }
    return r;
}

static __device__ __forceinline__ f32x8 wmma_bf16(bf16x16 a, bf16x16 b, f32x8 c) {
    // D = A(16x32 bf16) * B(32x16 bf16) + C(16x16 f32)
    return __builtin_amdgcn_wmma_f32_16x16x32_bf16(false, a, false, b, (short)0, c,
                                                   false, false);
}

// A-operand (16x32, MxK): lane<16 -> row, K = {k..k+7, k+16..k+23}
//                          lane>=16 -> row, K = {k+8..k+15, k+24..k+31}
static __device__ __forceinline__ bf16x16 loadA(const __bf16* row, int k, int half) {
    bf16x8 lo = *(const bf16x8*)(row + k + half * 8);
    bf16x8 hi = *(const bf16x8*)(row + k + 16 + half * 8);
    return cat8(lo, hi);
}

// B-operand (32x16, KxN): lane<16 -> col, K = {k..k+7, k+16..k+23} ; lane>=16 -> K = {k+8..k+15, k+24..k+31}
// (requires K-contiguous storage of the "column", i.e. transposed matrix)
static __device__ __forceinline__ bf16x16 loadB(const __bf16* col, int k, int half) {
    bf16x8 lo = *(const bf16x8*)(col + k + half * 8);
    bf16x8 hi = *(const bf16x8*)(col + k + 16 + half * 8);
    return cat8(lo, hi);
}

// Explicit 2-stage ping-pong 2x2-tile GEMM core over K = D_MODEL.
// Distinct X/Y operand buffer sets: no register rotation moves, and operand
// registers of an issued WMMA are not rewritten until a full stage later
// (avoids the 4-coexec WMMA->VALU WAR hazard NOPs).
// acc[0]=(m0,n0) acc[1]=(m0,n1) acc[2]=(m1,n0) acc[3]=(m1,n1)
static __device__ __forceinline__ void gemm2x2(
    const __bf16* __restrict__ arow0, const __bf16* __restrict__ arow1,
    const __bf16* __restrict__ wcol0, const __bf16* __restrict__ wcol1,
    int half, f32x8 acc[4]) {
    bf16x16 xa0 = loadA(arow0, 0, half), xa1 = loadA(arow1, 0, half);
    bf16x16 xb0 = loadB(wcol0, 0, half), xb1 = loadB(wcol1, 0, half);
    // chunks 1..30 in pairs; X holds even chunks, Y holds odd chunks
    for (int k = 32; k <= D_MODEL - 96; k += 64) {
        bf16x16 ya0 = loadA(arow0, k, half), ya1 = loadA(arow1, k, half);
        bf16x16 yb0 = loadB(wcol0, k, half), yb1 = loadB(wcol1, k, half);
        acc[0] = wmma_bf16(xa0, xb0, acc[0]);
        acc[1] = wmma_bf16(xa0, xb1, acc[1]);
        acc[2] = wmma_bf16(xa1, xb0, acc[2]);
        acc[3] = wmma_bf16(xa1, xb1, acc[3]);
        xa0 = loadA(arow0, k + 32, half); xa1 = loadA(arow1, k + 32, half);
        xb0 = loadB(wcol0, k + 32, half); xb1 = loadB(wcol1, k + 32, half);
        acc[0] = wmma_bf16(ya0, yb0, acc[0]);
        acc[1] = wmma_bf16(ya0, yb1, acc[1]);
        acc[2] = wmma_bf16(ya1, yb0, acc[2]);
        acc[3] = wmma_bf16(ya1, yb1, acc[3]);
    }
    // last chunk (k = D_MODEL-32) + drain
    {
        bf16x16 ya0 = loadA(arow0, D_MODEL - 32, half);
        bf16x16 ya1 = loadA(arow1, D_MODEL - 32, half);
        bf16x16 yb0 = loadB(wcol0, D_MODEL - 32, half);
        bf16x16 yb1 = loadB(wcol1, D_MODEL - 32, half);
        acc[0] = wmma_bf16(xa0, xb0, acc[0]);
        acc[1] = wmma_bf16(xa0, xb1, acc[1]);
        acc[2] = wmma_bf16(xa1, xb0, acc[2]);
        acc[3] = wmma_bf16(xa1, xb1, acc[3]);
        acc[0] = wmma_bf16(ya0, yb0, acc[0]);
        acc[1] = wmma_bf16(ya0, yb1, acc[1]);
        acc[2] = wmma_bf16(ya1, yb0, acc[2]);
        acc[3] = wmma_bf16(ya1, yb1, acc[3]);
    }
}

// ---------------- conversion kernels ----------------
__global__ void f32_to_bf16_kernel(const float* __restrict__ src,
                                   __bf16* __restrict__ dst, int n) {
    int i = blockIdx.x * 256 + threadIdx.x;
    if (i < n) dst[i] = (__bf16)src[i];
}

// Wt[n*1024 + k] = W[k*1024 + n]  (convert + transpose all 4 weights; z selects)
__global__ void w_transpose_kernel(const float* __restrict__ W0,
                                   const float* __restrict__ W1,
                                   const float* __restrict__ W2,
                                   const float* __restrict__ W3,
                                   __bf16* __restrict__ Wt) {
    int z = blockIdx.z;
    const float* W = (z == 0) ? W0 : (z == 1) ? W1 : (z == 2) ? W2 : W3;
    __bf16* Dst = Wt + (size_t)z * D_MODEL * D_MODEL;
    int id = blockIdx.x * 256 + threadIdx.x;          // 1M threads
    int k = id & (D_MODEL - 1);
    int n = id >> 10;
    Dst[(size_t)n * D_MODEL + k] = (__bf16)W[(size_t)k * D_MODEL + n];
}

// ---------------- fused QKV projection GEMM ----------------
// C[8192,1024] = xb @ W + b ; z = 0/1/2 -> Q/K/V
// Q written head-major [b,h,s,dk] with 1/sqrt(dk) folded in; K head-major;
// V head-transposed [b,h,dk,s].
// Block = 256 threads (8 waves), tile 128(M) x 64(N), 2x2 16x16 tiles per wave.
__global__ __launch_bounds__(256) void qkv_gemm_kernel(
    const __bf16* __restrict__ xb, const __bf16* __restrict__ Wt_all,
    const float* __restrict__ bq, const float* __restrict__ bk,
    const float* __restrict__ bv,
    __bf16* __restrict__ Qb, __bf16* __restrict__ Kb, __bf16* __restrict__ Vt) {
    int z = blockIdx.z;
    const __bf16* W   = Wt_all + (size_t)z * D_MODEL * D_MODEL;  // [N,K] transposed
    const float* bias = (z == 0) ? bq : (z == 1) ? bk : bv;
    float posc = (z == 0) ? 0.125f : 1.0f;          // fold 1/sqrt(64) into Q

    int w = threadIdx.x >> 5, lane = threadIdx.x & 31;
    int lr = lane & 15, half = lane >> 4;
    int mg = w & 3, ng = w >> 2;
    int m0 = blockIdx.y * 128 + mg * 32;            // wave's two m-tiles
    int m1 = m0 + 16;
    int nc0 = blockIdx.x * 64 + ng * 32 + lr;       // wave's two n-columns (this lane)
    int nc1 = nc0 + 16;

    const __bf16* arow0 = xb + (size_t)(m0 + lr) * D_MODEL;
    const __bf16* arow1 = xb + (size_t)(m1 + lr) * D_MODEL;
    const __bf16* wcol0 = W + (size_t)nc0 * D_MODEL;
    const __bf16* wcol1 = W + (size_t)nc1 * D_MODEL;

    f32x8 acc[4] = {};
    gemm2x2(arow0, arow1, wcol0, wcol1, half, acc);

#pragma unroll
    for (int i = 0; i < 2; ++i) {
        int mbase = i ? m1 : m0;
#pragma unroll
        for (int jn = 0; jn < 2; ++jn) {
            int n = jn ? nc1 : nc0;
            float bias_v = bias[n];
            int h = n >> 6, dk = n & 63;
            f32x8 a = acc[i * 2 + jn];
#pragma unroll
            for (int j = 0; j < 8; ++j) {
                int m = mbase + half * 8 + j;        // global token index
                int bb = m >> 11, s = m & (SEQ - 1);
                float v = (a[j] + bias_v) * posc;
                size_t bh = (size_t)bb * N_HEADS + h;
                if (z == 0)      Qb[(bh * SEQ + s) * D_K + dk] = (__bf16)v;
                else if (z == 1) Kb[(bh * SEQ + s) * D_K + dk] = (__bf16)v;
                else             Vt[(bh * D_K + dk) * SEQ + s] = (__bf16)v;
            }
        }
    }
}

// ---------------- flash attention ----------------
// grid: (SEQ/64, BATCH*N_HEADS); block: 256 threads (8 waves).
// Each block: one (b,h), 64 query rows. Loop over keys in chunks of 64 with
// online softmax; P@V accumulated in registers. 1/sqrt(dk) pre-folded into Q.
__global__ __launch_bounds__(256) void attn_kernel(
    const __bf16* __restrict__ Qb, const __bf16* __restrict__ Kb,
    const __bf16* __restrict__ Vt, __bf16* __restrict__ ctx) {
    __shared__ float  sS[64][65];       // fp32 score tile
    __shared__ __bf16 sP[64][72];       // bf16 probabilities (row stride 144B, 16B-aligned)
    __shared__ float  row_max[64], row_sum[64], row_scale[64];
    __shared__ float  pmax[64][4], psum[64][4];

    int bh = blockIdx.y;
    int q0 = blockIdx.x * 64;
    const __bf16* Qp = Qb + (size_t)bh * SEQ * D_K;
    const __bf16* Kp = Kb + (size_t)bh * SEQ * D_K;
    const __bf16* Vp = Vt + (size_t)bh * D_K * SEQ;

    int w = threadIdx.x >> 5, lane = threadIdx.x & 31;
    int lr = lane & 15, half = lane >> 4;
    int mtile = w & 3, grp = w >> 2;    // score n-tiles & O dk-tiles: grp*2 + {0,1}
    int rbase = mtile * 16 + half * 8;  // accumulator rows: rbase + j

    if (threadIdx.x < 64) { row_max[threadIdx.x] = -1e30f; row_sum[threadIdx.x] = 0.f; }

    // Q operands: reused across every key chunk (rows K-contiguous, dk = 64)
    const __bf16* qrow = Qp + (size_t)(q0 + mtile * 16 + lr) * D_K;
    bf16x16 qa0 = loadA(qrow, 0, half);
    bf16x16 qa1 = loadA(qrow, 32, half);

    // per-wave K rows (B-op columns) and V rows for its two tiles
    const __bf16* krow0 = Kp + (size_t)((grp * 2 + 0) * 16 + lr) * D_K;
    const __bf16* krow1 = Kp + (size_t)((grp * 2 + 1) * 16 + lr) * D_K;
    const __bf16* vrow0 = Vp + (size_t)((grp * 2 + 0) * 16 + lr) * SEQ;
    const __bf16* vrow1 = Vp + (size_t)((grp * 2 + 1) * 16 + lr) * SEQ;

    f32x8 o0 = {}, o1 = {};
    __syncthreads();

    for (int kc = 0; kc < SEQ; kc += 64) {
        // prefetch next chunk's K/V cachelines (global_prefetch_b8)
        if (kc + 64 < SEQ) {
            __builtin_prefetch(krow0 + (size_t)(kc + 64) * D_K, 0, 1);
            __builtin_prefetch(krow1 + (size_t)(kc + 64) * D_K, 0, 1);
            __builtin_prefetch(vrow0 + kc + 64, 0, 1);
            __builtin_prefetch(vrow1 + kc + 64, 0, 1);
        }

        // ---- scores: S = Qs @ K^T  (all loads hoisted ahead of the wmma group) ----
        {
            size_t ko = (size_t)kc * D_K;
            bf16x16 kb00 = loadB(krow0 + ko, 0, half);
            bf16x16 kb01 = loadB(krow0 + ko, 32, half);
            bf16x16 kb10 = loadB(krow1 + ko, 0, half);
            bf16x16 kb11 = loadB(krow1 + ko, 32, half);
            f32x8 s0 = {}, s1 = {};
            s0 = wmma_bf16(qa0, kb00, s0);
            s1 = wmma_bf16(qa0, kb10, s1);
            s0 = wmma_bf16(qa1, kb01, s0);
            s1 = wmma_bf16(qa1, kb11, s1);
#pragma unroll
            for (int j = 0; j < 8; ++j) {
                sS[rbase + j][(grp * 2 + 0) * 16 + lr] = s0[j];
                sS[rbase + j][(grp * 2 + 1) * 16 + lr] = s1[j];
            }
        }
        __syncthreads();

        // ---- online softmax: 4 threads per row, 16 columns each ----
        {
            int r = threadIdx.x >> 2, qd = threadIdx.x & 3;
            const float* srow = &sS[r][qd * 16];
            float pm = srow[0];
#pragma unroll
            for (int c = 1; c < 16; ++c) pm = fmaxf(pm, srow[c]);
            pmax[r][qd] = pm;
            __syncthreads();

            float om = row_max[r];
            float cm = fmaxf(fmaxf(fmaxf(pmax[r][0], pmax[r][1]),
                                   fmaxf(pmax[r][2], pmax[r][3])), om);
            float ps = 0.f;
#pragma unroll
            for (int c = 0; c < 16; ++c) {
                float p = __expf(srow[c] - cm);
                ps += p;
                sP[r][qd * 16 + c] = (__bf16)p;
            }
            psum[r][qd] = ps;
            __syncthreads();

            if (qd == 0) {
                float sc = __expf(om - cm);
                row_scale[r] = sc;
                row_max[r]   = cm;
                row_sum[r]   = row_sum[r] * sc +
                               psum[r][0] + psum[r][1] + psum[r][2] + psum[r][3];
            }
        }
        __syncthreads();

        // ---- rescale accumulators ----
#pragma unroll
        for (int j = 0; j < 8; ++j) {
            float sc = row_scale[rbase + j];
            o0[j] *= sc; o1[j] *= sc;
        }

        // ---- O += P @ V  (V stored [dk, s]: B-op columns are K-contiguous) ----
        {
            const __bf16* prow = &sP[mtile * 16 + lr][0];
            bf16x8 l0 = *(const bf16x8*)(prow + half * 8);
            bf16x8 h0 = *(const bf16x8*)(prow + 16 + half * 8);
            bf16x8 l1 = *(const bf16x8*)(prow + 32 + half * 8);
            bf16x8 h1 = *(const bf16x8*)(prow + 48 + half * 8);
            bf16x16 pa0 = cat8(l0, h0), pa1 = cat8(l1, h1);
            bf16x16 vb00 = loadB(vrow0 + kc, 0, half);
            bf16x16 vb01 = loadB(vrow0 + kc, 32, half);
            bf16x16 vb10 = loadB(vrow1 + kc, 0, half);
            bf16x16 vb11 = loadB(vrow1 + kc, 32, half);
            o0 = wmma_bf16(pa0, vb00, o0);
            o1 = wmma_bf16(pa0, vb10, o1);
            o0 = wmma_bf16(pa1, vb01, o0);
            o1 = wmma_bf16(pa1, vb11, o1);
        }
        __syncthreads();   // sS/sP reused next chunk
    }

    // ---- normalize and write context (token-major bf16) ----
    int b = bh >> 4, h = bh & 15;
#pragma unroll
    for (int j = 0; j < 8; ++j) {
        int rloc = rbase + j;
        float inv = 1.0f / row_sum[rloc];
        size_t base = ((size_t)(b * SEQ + q0 + rloc)) * D_MODEL + h * D_K;
        ctx[base + (grp * 2 + 0) * 16 + lr] = (__bf16)(o0[j] * inv);
        ctx[base + (grp * 2 + 1) * 16 + lr] = (__bf16)(o1[j] * inv);
    }
}

// ---------------- output projection: out = ctx @ Wo + bo (fp32 out) ----------------
__global__ __launch_bounds__(256) void out_gemm_kernel(
    const __bf16* __restrict__ ctx, const __bf16* __restrict__ Wot,
    const float* __restrict__ bo, float* __restrict__ out) {
    int w = threadIdx.x >> 5, lane = threadIdx.x & 31;
    int lr = lane & 15, half = lane >> 4;
    int mg = w & 3, ng = w >> 2;
    int m0 = blockIdx.y * 128 + mg * 32;
    int m1 = m0 + 16;
    int nc0 = blockIdx.x * 64 + ng * 32 + lr;
    int nc1 = nc0 + 16;

    const __bf16* arow0 = ctx + (size_t)(m0 + lr) * D_MODEL;
    const __bf16* arow1 = ctx + (size_t)(m1 + lr) * D_MODEL;
    const __bf16* wcol0 = Wot + (size_t)nc0 * D_MODEL;
    const __bf16* wcol1 = Wot + (size_t)nc1 * D_MODEL;

    f32x8 acc[4] = {};
    gemm2x2(arow0, arow1, wcol0, wcol1, half, acc);

#pragma unroll
    for (int i = 0; i < 2; ++i) {
        int mbase = i ? m1 : m0;
#pragma unroll
        for (int jn = 0; jn < 2; ++jn) {
            int n = jn ? nc1 : nc0;
            float bias_v = bo[n];
            f32x8 a = acc[i * 2 + jn];
#pragma unroll
            for (int j = 0; j < 8; ++j) {
                int m = mbase + half * 8 + j;
                out[(size_t)m * D_MODEL + n] = a[j] + bias_v;
            }
        }
    }
}

// ---------------- host entry ----------------
extern "C" void kernel_launch(void* const* d_in, const int* in_sizes, int n_in,
                              void* d_out, int out_size, void* d_ws, size_t ws_size,
                              hipStream_t stream) {
    const float* x  = (const float*)d_in[0];
    const float* Wq = (const float*)d_in[1];
    const float* bq = (const float*)d_in[2];
    const float* Wk = (const float*)d_in[3];
    const float* bk = (const float*)d_in[4];
    const float* Wv = (const float*)d_in[5];
    const float* bv = (const float*)d_in[6];
    const float* Wo = (const float*)d_in[7];
    const float* bo = (const float*)d_in[8];
    float* out = (float*)d_out;

    // workspace layout (bytes); total = 92,274,688 (~88 MB)
    char* ws = (char*)d_ws;
    __bf16* xb  = (__bf16*)(ws);                                  // 16 MB
    __bf16* Wt  = (__bf16*)(ws + (size_t)16777216);               //  8 MB (4 transposed weights)
    __bf16* Qb  = (__bf16*)(ws + (size_t)25165824);               // 16 MB  [b,h,s,dk] (pre-scaled)
    __bf16* Kb  = (__bf16*)(ws + (size_t)41943040);               // 16 MB  [b,h,s,dk]
    __bf16* Vt  = (__bf16*)(ws + (size_t)58720256);               // 16 MB  [b,h,dk,s]
    __bf16* ctx = (__bf16*)(ws + (size_t)75497472);               // 16 MB  [b,s,d]

    int n_x = in_sizes[0];                                        // 8,388,608
    f32_to_bf16_kernel<<<(n_x + 255) / 256, 256, 0, stream>>>(x, xb, n_x);

    w_transpose_kernel<<<dim3(D_MODEL * D_MODEL / 256, 1, 4), 256, 0, stream>>>(
        Wq, Wk, Wv, Wo, Wt);

    qkv_gemm_kernel<<<dim3(D_MODEL / 64, M_TOK / 128, 3), 256, 0, stream>>>(
        xb, Wt, bq, bk, bv, Qb, Kb, Vt);

    attn_kernel<<<dim3(SEQ / 64, BATCH * N_HEADS), 256, 0, stream>>>(Qb, Kb, Vt, ctx);

    out_gemm_kernel<<<dim3(D_MODEL / 64, M_TOK / 128), 256, 0, stream>>>(
        ctx, Wt + (size_t)3 * D_MODEL * D_MODEL, bo, out);
}
